// Pipeline_13572096656017
// MI455X (gfx1250) — compile-verified, matrix operand order checked by reading, not run on hardware
//
#include <hip/hip_runtime.h>
#include <hip/hip_bf16.h>

// ---------------- problem constants ----------------
#define B_ 64
#define N_ 1024
#define E_ 16384
#define F_ 128
#define H_ 64
#define T_ 10
#define C_ 2
#define NEG_ 0.01f
// sigmoid(z) >= p  <=>  z >= log(p/(1-p))
#define LOGIT_INNER (-0.8472978603872037f)   // log(0.3/0.7)
#define LOGIT_CROSS (-2.1972245773362196f)   // log(0.1/0.9)

static constexpr long BN_ = (long)B_ * N_;   // 65536
static constexpr long BE_ = (long)B_ * E_;   // 1048576

// ---------------- WMMA vector types (gfx1250, wave32) ----------------
typedef __attribute__((ext_vector_type(16))) __bf16 v16bf;
typedef __attribute__((ext_vector_type(8)))  float  v8f;

// ---------------- workspace layout (bytes) ----------------
static constexpr size_t OFF_INDEG = 0;                       // BN f32  (256 KB)
static constexpr size_t OFF_INVN  = 0x40000;                 // BN f32
static constexpr size_t OFF_MASK  = 0x80000;                 // BN u32
static constexpr size_t OFF_ISTOK = 0xC0000;                 // 16 f32
static constexpr size_t OFF_HTL1  = 0xC1000;                 // T*H f32
static constexpr size_t OFF_HTL2  = 0xC2000;                 // T*H f32
static constexpr size_t OFF_TSUM  = 0xC3000;                 // H f32
static constexpr size_t OFF_HNLIN = 0x100000;                // BN*H f32 (16.7 MB)
static constexpr size_t OFF_AGG   = OFF_HNLIN + (size_t)BN_ * H_ * 4;
static constexpr size_t OFF_H1    = OFF_AGG   + (size_t)BN_ * H_ * 4;
// total ~49.3 MB

// ---------------- CDNA5 async global->LDS copy (ASYNCcnt path) ----------
// 16B per issue; LDS byte address = low 32 bits of the generic pointer.
__device__ __forceinline__ void async_copy_b128(void* lds_dst, const void* gsrc) {
  const unsigned ldsa = (unsigned)(size_t)lds_dst;
  asm volatile("global_load_async_to_lds_b128 %0, %1, off"
               :: "v"(ldsa), "v"(gsrc) : "memory");
}
__device__ __forceinline__ void wait_asynccnt0() {
  asm volatile("s_wait_asynccnt 0" ::: "memory");
}

// =====================================================================
// Token-stream kernel: everything that only touches the T=10 tokens.
// =====================================================================
__global__ void tok_kernel(const float* __restrict__ tokens,
                           const float* __restrict__ W1, const float* __restrict__ b1,
                           const float* __restrict__ W2, const float* __restrict__ b2,
                           float* __restrict__ inv_s_tok,
                           float* __restrict__ ht_lin1, float* __restrict__ ht_lin2,
                           float* __restrict__ token_sum) {
  __shared__ float tok[T_][F_];
  __shared__ float minf[T_][T_];
  __shared__ float degt[T_];
  __shared__ float htl1[T_][H_];
  __shared__ float ht1a[T_][H_];
  __shared__ float htl2[T_][H_];
  const int tid = threadIdx.x;           // 64 threads (2 waves)
  for (int i = tid; i < T_ * F_; i += 64) tok[i / F_][i % F_] = tokens[i];
  __syncthreads();
  if (tid < T_) {
    for (int u = 0; u < T_; ++u) {
      float s = 0.f;
      for (int f = 0; f < F_; ++f) s += tok[tid][f] * tok[u][f];
      minf[tid][u] = (s >= LOGIT_INNER) ? 1.f : 0.f;
    }
  }
  __syncthreads();
  if (tid < T_) {
    float d = 1.f;
    for (int t = 0; t < T_; ++t) d += minf[t][tid];   // M_in.sum(0)
    degt[tid] = d;
    inv_s_tok[tid] = rsqrtf(d);
  }
  __syncthreads();
  if (tid < H_) {
    for (int t = 0; t < T_; ++t) {                     // ht_lin1 = tokens @ W1
      float s = 0.f;
      for (int f = 0; f < F_; ++f) s += tok[t][f] * W1[f * H_ + tid];
      htl1[t][tid] = s;
      ht_lin1[t * H_ + tid] = s;
    }
  }
  __syncthreads();
  if (tid < H_) {
    for (int t = 0; t < T_; ++t) {
      const float ist = rsqrtf(degt[t]);
      float s = 0.f;
      for (int u = 0; u < T_; ++u)
        s += minf[t][u] * ist * rsqrtf(degt[u]) * htl1[u][tid];  // norm_in @ ht
      s += htl1[t][tid] / degt[t] + b1[tid];
      ht1a[t][tid] = (s >= 0.f) ? s : NEG_ * s;        // leaky relu
    }
  }
  __syncthreads();
  if (tid < H_) {
    for (int t = 0; t < T_; ++t) {                     // ht_lin2 = ht1a @ W2
      float s = 0.f;
      for (int k = 0; k < H_; ++k) s += ht1a[t][k] * W2[k * H_ + tid];
      htl2[t][tid] = s;
      ht_lin2[t * H_ + tid] = s;
    }
  }
  __syncthreads();
  if (tid < H_) {
    float acc = 0.f;
    for (int t = 0; t < T_; ++t) {
      const float ist = rsqrtf(degt[t]);
      float s = 0.f;
      for (int u = 0; u < T_; ++u)
        s += minf[t][u] * ist * rsqrtf(degt[u]) * htl2[u][tid];
      s += htl2[t][tid] / degt[t] + b2[tid];
      acc += s;                                        // ht.sum(0) for pooling
    }
    token_sum[tid] = acc;
  }
}

// =====================================================================
// M_cr as a 10-bit mask per node: bit t set iff sigmoid(tok_t . x_bn) >= 0.1
// Token tile staged via CDNA5 async global->LDS (ASYNCcnt).
// =====================================================================
__global__ void mcr_kernel(const float* __restrict__ x,
                           const float* __restrict__ tokens,
                           unsigned* __restrict__ mask) {
  __shared__ float tok[T_][F_];
  const int tid = threadIdx.x;
  constexpr int NVEC = T_ * F_ / 4;                    // 320 x 16B chunks
  for (int i = tid; i < NVEC; i += 256)
    async_copy_b128(&tok[0][0] + i * 4, tokens + i * 4);
  wait_asynccnt0();
  __syncthreads();
  const long i = (long)blockIdx.x * 256 + tid;         // global node id in [0,BN)
  const float4* xp = (const float4*)(x + i * F_);
  float acc[T_];
#pragma unroll
  for (int t = 0; t < T_; ++t) acc[t] = 0.f;
  for (int f = 0; f < F_; f += 4) {
    const float4 xv = xp[f >> 2];
#pragma unroll
    for (int t = 0; t < T_; ++t)
      acc[t] += xv.x * tok[t][f] + xv.y * tok[t][f + 1] +
                xv.z * tok[t][f + 2] + xv.w * tok[t][f + 3];
  }
  unsigned m = 0;
#pragma unroll
  for (int t = 0; t < T_; ++t)
    if (acc[t] >= LOGIT_CROSS) m |= (1u << t);
  mask[i] = m;
}

// ---------------- edge in-degree (atomics into f32) ----------------
__global__ void edge_deg_kernel(const int* __restrict__ edst,
                                float* __restrict__ indeg) {
  const long e = (long)blockIdx.x * 256 + threadIdx.x;  // [0, B*E)
  const int b = (int)(e >> 14);                         // E = 16384 = 2^14
  atomicAdd(&indeg[(long)b * N_ + edst[e]], 1.0f);
}

// ---------------- deg_node = 1 + indeg + popcount(mask); store rsqrt ----
__global__ void deg_kernel(const float* __restrict__ indeg,
                           const unsigned* __restrict__ mask,
                           float* __restrict__ inv_node) {
  const long i = (long)blockIdx.x * 256 + threadIdx.x;
  const float d = 1.f + indeg[i] + (float)__popc(mask[i]);
  inv_node[i] = rsqrtf(d);
}

// =====================================================================
// WMMA GEMM: C[M,64] = A[M,K] @ Bm[K,64], f32 in/out, bf16 multiply.
// Block = 256 threads = 8 waves. Each wave computes a 16x64 strip: one A
// fragment feeds 4 consecutive v_wmma (one per 16-col tile), so A (the
// 33 MB x tensor) is read exactly once from HBM. W is pre-swizzled into
// LDS as bf16 already in the ISA 7.12.2 B-fragment layout (<=16 KB), so
// inner-loop B operands are two ds_load_b128 per tile.
// =====================================================================
template <int K>
__global__ void gemm_wmma_kernel(const float* __restrict__ A,
                                 const float* __restrict__ Bm,
                                 float* __restrict__ C) {
  constexpr int KS = K / 32;                           // 4 or 2 k-steps
  __shared__ v16bf Wlds[KS][4][32];                    // [kstep][colTile][lane]
  const int tid = threadIdx.x;

  // ---- pre-swizzle W (f32 [K,64]) -> bf16 B fragments in LDS ----
  constexpr int NC = KS * 4 * 32;
  for (int c = tid; c < NC; c += 256) {
    const int lc = c & 31;
    const int ct = (c >> 5) & 3;
    const int ks = c >> 7;
    const int base = (lc & 16) ? 8 : 0;
    const int col = ct * 16 + (lc & 15);
    v16bf bv;
#pragma unroll
    for (int j = 0; j < 16; ++j) {
      const int koff = base + j + ((j < 8) ? 0 : 8);   // {0..7,16..23} / {8..15,24..31}
      bv[j] = (__bf16)Bm[(ks * 32 + koff) * H_ + col];
    }
    Wlds[ks][ct][lc] = bv;
  }
  __syncthreads();

  const int wave = tid >> 5;
  const int lane = tid & 31;
  const int l15  = lane & 15;
  const int hi   = lane >> 4;
  const int kb   = hi * 8;
  const long rowBase = (long)blockIdx.x * 128 + wave * 16;
  const float* arow = A + (rowBase + l15) * K;

  v8f acc[4] = {{}, {}, {}, {}};
#pragma unroll
  for (int ks = 0; ks < KS; ++ks) {
    const float4* pa = (const float4*)(arow + ks * 32 + kb);
    const float4 f0 = pa[0], f1 = pa[1];               // K = +kb .. +kb+7
    const float4 f2 = pa[4], f3 = pa[5];               // K = +kb+16 .. +kb+23
    v16bf av;
    av[0]  = (__bf16)f0.x; av[1]  = (__bf16)f0.y; av[2]  = (__bf16)f0.z; av[3]  = (__bf16)f0.w;
    av[4]  = (__bf16)f1.x; av[5]  = (__bf16)f1.y; av[6]  = (__bf16)f1.z; av[7]  = (__bf16)f1.w;
    av[8]  = (__bf16)f2.x; av[9]  = (__bf16)f2.y; av[10] = (__bf16)f2.z; av[11] = (__bf16)f2.w;
    av[12] = (__bf16)f3.x; av[13] = (__bf16)f3.y; av[14] = (__bf16)f3.z; av[15] = (__bf16)f3.w;
#pragma unroll
    for (int ct = 0; ct < 4; ++ct) {
      acc[ct] = __builtin_amdgcn_wmma_f32_16x16x32_bf16(
          /*neg_a=*/false, av, /*neg_b=*/false, Wlds[ks][ct][lane],
          /*c_mod=*/(short)0, acc[ct], /*reuse_a=*/false, /*reuse_b=*/false);
    }
  }
  // ---- C store: VGPR r -> row rowBase + r + 8*hi, col ct*16 + l15 ----
#pragma unroll
  for (int ct = 0; ct < 4; ++ct) {
#pragma unroll
    for (int r = 0; r < 8; ++r) {
      const long orow = rowBase + r + hi * 8;
      C[orow * H_ + ct * 16 + l15] = acc[ct][r];
    }
  }
}

// ---------------- edge scatter: agg[dst] += hn[src] * ew ----------------
__global__ void scatter_kernel(const int* __restrict__ esrc,
                               const int* __restrict__ edst,
                               const float* __restrict__ inv_node,
                               const float* __restrict__ hn,
                               float* __restrict__ agg) {
  const long e = (long)blockIdx.x * 256 + threadIdx.x;
  const int b = (int)(e >> 14);
  const long is = (long)b * N_ + esrc[e];
  const long id = (long)b * N_ + edst[e];
  const float w = inv_node[is] * inv_node[id];
  const float4* s = (const float4*)(hn + is * H_);
  float* d = agg + id * H_;
#pragma unroll 4
  for (int k = 0; k < H_ / 4; ++k) {
    const float4 v = s[k];
    atomicAdd(&d[k * 4 + 0], v.x * w);
    atomicAdd(&d[k * 4 + 1], v.y * w);
    atomicAdd(&d[k * 4 + 2], v.z * w);
    atomicAdd(&d[k * 4 + 3], v.w * w);
  }
}

// ---------------- fuse: out = agg + cross + self + bias (opt. lrelu) -----
__global__ void fuse_kernel(const float* __restrict__ hn,
                            const float* __restrict__ agg,
                            const unsigned* __restrict__ mask,
                            const float* __restrict__ inv_node,
                            const float* __restrict__ inv_s_tok,
                            const float* __restrict__ htl,
                            const float* __restrict__ bias,
                            float* __restrict__ out, int do_act) {
  const long idx = (long)blockIdx.x * 256 + threadIdx.x;  // [0, BN*H)
  const long i = idx >> 6;
  const int o = (int)(idx & 63);
  const unsigned m = mask[i];
  float cross = 0.f;
#pragma unroll
  for (int t = 0; t < T_; ++t)
    if (m & (1u << t)) cross += inv_s_tok[t] * htl[t * H_ + o];
  const float isn = inv_node[i];
  float v = agg[idx] + cross * isn + hn[idx] * (isn * isn) + bias[o];
  if (do_act) v = (v >= 0.f) ? v : NEG_ * v;
  out[idx] = v;
}

// ---------------- per-graph mean pool + answering head + softmax --------
__global__ void pool_head_kernel(const float* __restrict__ h2,
                                 const float* __restrict__ token_sum,
                                 const float* __restrict__ Wa,
                                 const float* __restrict__ ba,
                                 float* __restrict__ out) {
  __shared__ float sd[256];
  __shared__ float ge[H_];
  const int b = blockIdx.x;
  const int tid = threadIdx.x;
  const int col = tid & 63, part = tid >> 6;
  float s = 0.f;
  for (int n = part; n < N_; n += 4)
    s += h2[((long)b * N_ + n) * H_ + col];
  sd[tid] = s;
  __syncthreads();
  if (part == 0) {
    const float tot = sd[col] + sd[64 + col] + sd[128 + col] + sd[192 + col];
    ge[col] = (token_sum[col] + tot) * (1.0f / (float)(T_ + N_));
  }
  __syncthreads();
  if (tid == 0) {
    float l0 = ba[0], l1 = ba[1];
    for (int k = 0; k < H_; ++k) {
      l0 += ge[k] * Wa[k * C_ + 0];
      l1 += ge[k] * Wa[k * C_ + 1];
    }
    const float mx = fmaxf(l0, l1);
    const float e0 = __expf(l0 - mx), e1 = __expf(l1 - mx);
    const float inv = 1.f / (e0 + e1);
    out[b * C_ + 0] = e0 * inv;
    out[b * C_ + 1] = e1 * inv;
  }
}

// =====================================================================
extern "C" void kernel_launch(void* const* d_in, const int* in_sizes, int n_in,
                              void* d_out, int out_size, void* d_ws, size_t ws_size,
                              hipStream_t stream) {
  (void)in_sizes; (void)n_in; (void)out_size; (void)ws_size;
  const float* x      = (const float*)d_in[0];
  const float* tokens = (const float*)d_in[1];
  const float* W1     = (const float*)d_in[2];
  const float* b1     = (const float*)d_in[3];
  const float* W2     = (const float*)d_in[4];
  const float* b2     = (const float*)d_in[5];
  const float* Wa     = (const float*)d_in[6];
  const float* ba     = (const float*)d_in[7];
  const int* esrc     = (const int*)d_in[8];
  const int* edst     = (const int*)d_in[9];
  float* out          = (float*)d_out;

  char* ws = (char*)d_ws;
  float*    indeg   = (float*)(ws + OFF_INDEG);
  float*    inv_node= (float*)(ws + OFF_INVN);
  unsigned* mask    = (unsigned*)(ws + OFF_MASK);
  float*    istok   = (float*)(ws + OFF_ISTOK);
  float*    htl1    = (float*)(ws + OFF_HTL1);
  float*    htl2    = (float*)(ws + OFF_HTL2);
  float*    tsum    = (float*)(ws + OFF_TSUM);
  float*    hn_lin  = (float*)(ws + OFF_HNLIN);
  float*    agg     = (float*)(ws + OFF_AGG);
  float*    h1      = (float*)(ws + OFF_H1);

  const int nodeBlocks = (int)(BN_ / 256);        // 256
  const int edgeBlocks = (int)(BE_ / 256);        // 4096
  const int gemmBlocks = (int)(BN_ / 128);        // 512  (128 rows / block)
  const int fuseBlocks = (int)(BN_ * H_ / 256);   // 16384

  // degree/prompt prep
  hipMemsetAsync(indeg, 0, (size_t)BN_ * 4, stream);
  tok_kernel<<<1, 64, 0, stream>>>(tokens, W1, b1, W2, b2, istok, htl1, htl2, tsum);
  mcr_kernel<<<nodeBlocks, 256, 0, stream>>>(x, tokens, mask);
  edge_deg_kernel<<<edgeBlocks, 256, 0, stream>>>(edst, indeg);
  deg_kernel<<<nodeBlocks, 256, 0, stream>>>(indeg, mask, inv_node);

  // ---- layer 1 ----
  gemm_wmma_kernel<F_><<<gemmBlocks, 256, 0, stream>>>(x, W1, hn_lin);
  hipMemsetAsync(agg, 0, (size_t)BN_ * H_ * 4, stream);
  scatter_kernel<<<edgeBlocks, 256, 0, stream>>>(esrc, edst, inv_node, hn_lin, agg);
  fuse_kernel<<<fuseBlocks, 256, 0, stream>>>(hn_lin, agg, mask, inv_node, istok,
                                              htl1, b1, h1, /*do_act=*/1);

  // ---- layer 2 ----
  gemm_wmma_kernel<H_><<<gemmBlocks, 256, 0, stream>>>(h1, W2, hn_lin);
  hipMemsetAsync(agg, 0, (size_t)BN_ * H_ * 4, stream);
  scatter_kernel<<<edgeBlocks, 256, 0, stream>>>(esrc, edst, inv_node, hn_lin, agg);
  fuse_kernel<<<fuseBlocks, 256, 0, stream>>>(hn_lin, agg, mask, inv_node, istok,
                                              htl2, b2, h1, /*do_act=*/0);

  // ---- pool + head + softmax ----
  pool_head_kernel<<<B_, 256, 0, stream>>>(h1, tsum, Wa, ba, out);
}
